// attention_block_66855460929923
// MI455X (gfx1250) — matrix operand order repaired and checked by reading, not compiled, not run on hardware
//
#include <hip/hip_runtime.h>

#define D_MODEL 1024
#define SEQ     4096
#define NBATCH  4
#define NROWS   (NBATCH * SEQ)   // 16384 flat rows

typedef __attribute__((ext_vector_type(16))) _Float16 v16h;
typedef __attribute__((ext_vector_type(8)))  _Float16 v8h;
typedef __attribute__((ext_vector_type(8)))  float    v8f;

// ---------------------------------------------------------------------------
// D = A(16x32 f16) * B(32x16 f16) + C(16x16 f32), wave32 WMMA
// ---------------------------------------------------------------------------
__device__ __forceinline__ v8f wmma16(v16h a, v16h b, v8f c) {
  return __builtin_amdgcn_wmma_f32_16x16x32_f16(
      /*neg_a=*/false, a, /*neg_b=*/false, b,
      /*c_mod=*/(short)0, c, /*reuse_a=*/false, /*reuse_b=*/false);
}

// A-matrix fragment (16 rows x 32 K), row-major source, stride ld halfs.
// Lanes 0-15 = rows, K packed [0..7 | 16..23]; lanes 16-31 same rows,
// K [8..15 | 24..31].  Two 16B loads per lane.
__device__ __forceinline__ v16h load_fragA(const _Float16* base, int ld,
                                           int k0, int lane) {
  const int r  = lane & 15;
  const int hi = (lane >> 4) & 1;
  const _Float16* p = base + (size_t)r * ld + k0 + hi * 8;
  v8h lo = *(const v8h*)(p);
  v8h hh = *(const v8h*)(p + 16);
  v16h o;
#pragma unroll
  for (int i = 0; i < 8; ++i) { o[i] = lo[i]; o[i + 8] = hh[i]; }
  return o;
}

// B-matrix fragment (32 K x 16 cols) given B^T row-major (row n = column n).
// Lane n holds column n; lanes 0-15 K [0..7 | 16..23], lanes 16-31 K [8..15 | 24..31].
__device__ __forceinline__ v16h load_fragB(const _Float16* baseT, int ld,
                                           int k0, int lane) {
  const int n  = lane & 15;
  const int hi = (lane >> 4) & 1;
  const _Float16* p = baseT + (size_t)n * ld + k0 + hi * 8;
  v8h lo = *(const v8h*)(p);
  v8h hh = *(const v8h*)(p + 16);
  v16h o;
#pragma unroll
  for (int i = 0; i < 8; ++i) { o[i] = lo[i]; o[i + 8] = hh[i]; }
  return o;
}

// ---------------------------------------------------------------------------
// CDNA5 async memory->LDS DMA (ASYNCcnt-tracked), inline asm per
// CDNA5 ISA.  lds_byte_off is the wave-relative LDS
// address (single __shared__ array -> allocated at LDS offset 0).
// ---------------------------------------------------------------------------
__device__ __forceinline__ void async_ld_b128(unsigned lds_byte_off,
                                              const void* gaddr) {
  asm volatile("global_load_async_to_lds_b128 %0, %1, off"
               :: "v"(lds_byte_off), "v"(gaddr)
               : "memory");
}
__device__ __forceinline__ void wait_async0() {
  asm volatile("s_wait_asynccnt 0x0" ::: "memory");
}

// ---------------------------------------------------------------------------
// Kernel 1: fp32 -> f16 pack (grid-stride)
// ---------------------------------------------------------------------------
__global__ void pack_f16_kernel(const float* __restrict__ src,
                                _Float16* __restrict__ dst, int n) {
  int i = blockIdx.x * blockDim.x + threadIdx.x;
  int stride = gridDim.x * blockDim.x;
  for (; i < n; i += stride) dst[i] = (_Float16)src[i];
}

// ---------------------------------------------------------------------------
// Kernel 2: QKV projection.  grid = (NROWS/64, D/128, 3), block = 128.
// Each wave computes a 32x64 tile (2x4 WMMA tiles): A frags reused across
// 4 B tiles -> 12 b128 loads per 8 WMMAs.
// mode 0/1 -> Q/K row-major f16 [row, e]; mode 2 -> Vt f16 [b][e][s].
// ---------------------------------------------------------------------------
__global__ __launch_bounds__(128)
void qkv_gemm_kernel(const _Float16* __restrict__ xh,
                     const _Float16* __restrict__ wh,   // [3][D][D] (e,d)
                     const float* __restrict__ bq,
                     const float* __restrict__ bk,
                     const float* __restrict__ bv,
                     _Float16* __restrict__ qh,
                     _Float16* __restrict__ kh,
                     _Float16* __restrict__ vth) {
  const int lane = threadIdx.x & 31;
  const int w    = threadIdx.x >> 5;
  const int hi   = (lane >> 4) & 1;
  const int mode = blockIdx.z;

  const _Float16* W = wh + (size_t)mode * D_MODEL * D_MODEL;
  const float* bias = (mode == 0) ? bq : (mode == 1) ? bk : bv;

  const int m0 = blockIdx.x * 64 + (w & 1) * 32;
  const int n0 = blockIdx.y * 128 + (w >> 1) * 64;

  v8f c[2][4] = {};
  for (int d0 = 0; d0 < D_MODEL; d0 += 32) {
    v16h a0 = load_fragA(xh + (size_t)m0 * D_MODEL, D_MODEL, d0, lane);
    v16h a1 = load_fragA(xh + (size_t)(m0 + 16) * D_MODEL, D_MODEL, d0, lane);
#pragma unroll
    for (int j = 0; j < 4; ++j) {
      v16h bj = load_fragB(W + (size_t)(n0 + j * 16) * D_MODEL, D_MODEL, d0, lane);
      c[0][j] = wmma16(a0, bj, c[0][j]);
      c[1][j] = wmma16(a1, bj, c[1][j]);
    }
  }

#pragma unroll
  for (int i = 0; i < 2; ++i) {
#pragma unroll
    for (int j = 0; j < 4; ++j) {
      const int col = n0 + j * 16 + (lane & 15);
      const float bsc = bias[col];
#pragma unroll
      for (int e = 0; e < 8; ++e) {
        const int row = m0 + i * 16 + e + hi * 8;   // C layout: VGPR e -> M
        const float val = c[i][j][e] + bsc;
        if (mode == 0) {
          qh[(size_t)row * D_MODEL + col] = (_Float16)val;
        } else if (mode == 1) {
          kh[(size_t)row * D_MODEL + col] = (_Float16)val;
        } else {
          const int bb = row >> 12, sl = row & (SEQ - 1);
          vth[((size_t)bb * D_MODEL + col) * SEQ + sl] = (_Float16)val;
        }
      }
    }
  }
}

// ---------------------------------------------------------------------------
// Kernel 3: scaled scores S = (Q K^T)/sqrt(D) written f16 + online row stats.
// grid = NROWS/32, block = 64 (2 waves).  Q block (32 x 1024 f16 = 64 KB)
// staged in LDS via global_load_async_to_lds_b128; K streamed from L2.
// Each wave owns a 16-row q-tile over the full k range and processes 4
// k-tiles per A fragment (10 loads / 4 WMMAs).
// ---------------------------------------------------------------------------
__global__ __launch_bounds__(64)
void scores_stats_kernel(const _Float16* __restrict__ qh,
                         const _Float16* __restrict__ kh,
                         _Float16* __restrict__ sh,
                         float* __restrict__ mrow,
                         float* __restrict__ lrow) {
  __shared__ _Float16 Qs[32 * D_MODEL];   // 64 KB, LDS offset 0
  const int tid  = threadIdx.x;
  const int lane = tid & 31;
  const int w    = tid >> 5;
  const int hi   = (lane >> 4) & 1;
  const int flat0 = blockIdx.x * 32;      // flat row (batch folded)
  const int b     = flat0 >> 12;

  // async DMA stage: 4096 x b128 chunks / 64 threads, memory -> LDS direct
  for (int it = 0; it < 64; ++it) {
    const int idx = tid + it * 64;
    const int r = idx >> 7;               // 128 chunks per row
    const int c8 = idx & 127;
    async_ld_b128((unsigned)((r * D_MODEL + c8 * 8) * (int)sizeof(_Float16)),
                  &qh[(size_t)(flat0 + r) * D_MODEL + c8 * 8]);
  }
  wait_async0();
  __syncthreads();

  const _Float16* qbase = &Qs[w * 16 * D_MODEL];
  const float scale = 0.03125f;           // 1/sqrt(1024)

  float rmax[8], rsum[8];
#pragma unroll
  for (int e = 0; e < 8; ++e) { rmax[e] = -1e30f; rsum[e] = 0.0f; }

  for (int kt = 0; kt < SEQ / 64; ++kt) { // 64 k-columns per iteration
    v8f c[4] = {};
    for (int d0 = 0; d0 < D_MODEL; d0 += 32) {
      v16h aq = load_fragA(qbase, D_MODEL, d0, lane);
#pragma unroll
      for (int t = 0; t < 4; ++t) {
        const _Float16* kbase =
            kh + (size_t)(b * SEQ + kt * 64 + t * 16) * D_MODEL;
        v16h bk2 = load_fragB(kbase, D_MODEL, d0, lane);
        c[t] = wmma16(aq, bk2, c[t]);
      }
    }
    const size_t srow0 = (size_t)(flat0 + w * 16 + hi * 8) * SEQ;
#pragma unroll
    for (int t = 0; t < 4; ++t) {
      const int col = kt * 64 + t * 16 + (lane & 15);
#pragma unroll
      for (int e = 0; e < 8; ++e) {
        const float v = c[t][e] * scale;
        sh[srow0 + (size_t)e * SEQ + col] = (_Float16)v;
        // row lives in one VGPR across a 16-lane half -> 4 xor-shuffles
        float tm = v;
        tm = fmaxf(tm, __shfl_xor(tm, 1, 32));
        tm = fmaxf(tm, __shfl_xor(tm, 2, 32));
        tm = fmaxf(tm, __shfl_xor(tm, 4, 32));
        tm = fmaxf(tm, __shfl_xor(tm, 8, 32));
        const float nm = fmaxf(rmax[e], tm);
        float ex = __expf(v - nm);
        ex += __shfl_xor(ex, 1, 32);
        ex += __shfl_xor(ex, 2, 32);
        ex += __shfl_xor(ex, 4, 32);
        ex += __shfl_xor(ex, 8, 32);
        rsum[e] = rsum[e] * __expf(rmax[e] - nm) + ex;
        rmax[e] = nm;
      }
    }
  }
  if ((lane & 15) == 0) {
#pragma unroll
    for (int e = 0; e < 8; ++e) {
      const int row = flat0 + w * 16 + hi * 8 + e;
      mrow[row] = rmax[e];
      lrow[row] = rsum[e];
    }
  }
}

// ---------------------------------------------------------------------------
// Kernel 4: normalize P in place, then O = P V via WMMA.
// grid = NROWS/64, block = 128 (4 waves); wave w owns q-tile w, loops 8
// e-chunks of 128 (8 WMMA accum tiles each, A frag reused 8x) over full k.
// ---------------------------------------------------------------------------
__global__ __launch_bounds__(128)
void attn_out_kernel(_Float16* __restrict__ sh,
                     const float* __restrict__ mrow,
                     const float* __restrict__ lrow,
                     const _Float16* __restrict__ vth,
                     float* __restrict__ out) {
  const int tid  = threadIdx.x;
  const int lane = tid & 31;
  const int w    = tid >> 5;
  const int hi   = (lane >> 4) & 1;
  const int flat0 = blockIdx.x * 64;
  const int b     = flat0 >> 12;

  // phase 0: P = exp(S - m) / l, in place on this block's 64-row strip
  for (int it = 0; it < 256; ++it) {
    const int idx = tid + it * 128;       // 32768 v8h chunks
    const int r  = idx >> 9;              // 512 chunks per row
    const int c8 = idx & 511;
    const int gr = flat0 + r;
    const float m    = mrow[gr];
    const float invl = 1.0f / lrow[gr];
    v8h s8 = *(v8h*)&sh[(size_t)gr * SEQ + c8 * 8];
#pragma unroll
    for (int i = 0; i < 8; ++i)
      s8[i] = (_Float16)(__expf((float)s8[i] - m) * invl);
    *(v8h*)&sh[(size_t)gr * SEQ + c8 * 8] = s8;
  }
  __threadfence();
  __syncthreads();

  // phase 1: O[16 x 1024] = P[16 x 4096] * V[4096 x 1024]  (Vt is B^T)
  const _Float16* pbase = sh + (size_t)(flat0 + w * 16) * SEQ;
  for (int ec = 0; ec < 8; ++ec) {
    v8f c[8] = {};
    for (int k0 = 0; k0 < SEQ; k0 += 32) {
      v16h ap = load_fragA(pbase, SEQ, k0, lane);
#pragma unroll
      for (int j = 0; j < 8; ++j) {
        v16h bv2 = load_fragB(
            vth + ((size_t)b * D_MODEL + ec * 128 + j * 16) * SEQ, SEQ, k0, lane);
        c[j] = wmma16(ap, bv2, c[j]);
      }
    }
#pragma unroll
    for (int j = 0; j < 8; ++j) {
      const int col = ec * 128 + j * 16 + (lane & 15);
#pragma unroll
      for (int e = 0; e < 8; ++e) {
        const int row = flat0 + w * 16 + e + hi * 8;
        out[(size_t)row * D_MODEL + col] = c[j][e];
      }
    }
  }
}

// ---------------------------------------------------------------------------
// Workspace layout (f16 = 2B):  x_h 32MB | W_h 6MB | Q_h 32MB | K_h 32MB |
// Vt_h 32MB | scores 128MB | m 64KB | l 64KB   (total ~275 MB)
// ---------------------------------------------------------------------------
extern "C" void kernel_launch(void* const* d_in, const int* in_sizes, int n_in,
                              void* d_out, int out_size, void* d_ws,
                              size_t ws_size, hipStream_t stream) {
  (void)in_sizes; (void)n_in; (void)out_size; (void)ws_size;
  const float* x  = (const float*)d_in[0];
  const float* Wq = (const float*)d_in[1];
  const float* bq = (const float*)d_in[2];
  const float* Wk = (const float*)d_in[3];
  const float* bk = (const float*)d_in[4];
  const float* Wv = (const float*)d_in[5];
  const float* bv = (const float*)d_in[6];
  float* out = (float*)d_out;

  char* ws = (char*)d_ws;
  size_t off = 0;
  auto alloc = [&](size_t bytes) -> char* {
    char* p = ws + off;
    off += (bytes + 255) & ~(size_t)255;
    return p;
  };
  _Float16* xh  = (_Float16*)alloc(sizeof(_Float16) * (size_t)NROWS * D_MODEL);
  _Float16* wh  = (_Float16*)alloc(sizeof(_Float16) * 3 * (size_t)D_MODEL * D_MODEL);
  _Float16* qh  = (_Float16*)alloc(sizeof(_Float16) * (size_t)NROWS * D_MODEL);
  _Float16* kh  = (_Float16*)alloc(sizeof(_Float16) * (size_t)NROWS * D_MODEL);
  _Float16* vth = (_Float16*)alloc(sizeof(_Float16) * (size_t)NROWS * D_MODEL);
  _Float16* sh  = (_Float16*)alloc(sizeof(_Float16) * (size_t)NROWS * SEQ);
  float*    mrow = (float*)alloc(sizeof(float) * NROWS);
  float*    lrow = (float*)alloc(sizeof(float) * NROWS);

  pack_f16_kernel<<<dim3(4096), dim3(256), 0, stream>>>(x, xh, NROWS * D_MODEL);
  pack_f16_kernel<<<dim3(1024), dim3(256), 0, stream>>>(Wq, wh, D_MODEL * D_MODEL);
  pack_f16_kernel<<<dim3(1024), dim3(256), 0, stream>>>(Wk, wh + (size_t)D_MODEL * D_MODEL, D_MODEL * D_MODEL);
  pack_f16_kernel<<<dim3(1024), dim3(256), 0, stream>>>(Wv, wh + 2 * (size_t)D_MODEL * D_MODEL, D_MODEL * D_MODEL);

  qkv_gemm_kernel<<<dim3(NROWS / 64, D_MODEL / 128, 3), dim3(128), 0, stream>>>(
      xh, wh, bq, bk, bv, qh, kh, vth);

  scores_stats_kernel<<<dim3(NROWS / 32), dim3(64), 0, stream>>>(
      qh, kh, sh, mrow, lrow);

  attn_out_kernel<<<dim3(NROWS / 64), dim3(128), 0, stream>>>(
      sh, mrow, lrow, vth, out);
}
